// MoE_74655121539086
// MI455X (gfx1250) — matrix operand order repaired and checked by reading, not compiled, not run on hardware
//
#include <hip/hip_runtime.h>
#include <hip/hip_bf16.h>

// ---------------------------------------------------------------------------
// MoE (top-2 of 16 routed experts + 2 shared experts) for MI455X / gfx1250.
// Compute-bound (~0.93 TFLOP, AI ~1650 FLOP/B) => bf16 WMMA with f32 accum.
// Round 3: GEMM staging converted to GLOBAL_LOAD_ASYNC_TO_LDS_B128 with
// double-buffered LDS + s_wait_asynccnt (ASYNCcnt) pipelining; inner loop is
// now: issue async copies -> wait -> barrier -> ds_load frags + 8 WMMA.
// Workspace requirement: ~429 MB.
// ---------------------------------------------------------------------------

typedef __attribute__((ext_vector_type(16))) __bf16 v16bf;
typedef __attribute__((ext_vector_type(8)))  float  v8f;

#define NTOK   16384   // B*T
#define DDIM   1024
#define HDIM   2048
#define NEXP   16
#define NSH    2
#define CAP    2560    // ceil(NTOK*2/16 * 1.25)
#define RSCALE 1.0f

// ------------------------- async copy primitives ---------------------------
// Per-lane DMA: LDS[vdst_lane] = MEM[vaddr_lane .. +15]; tracked by ASYNCcnt.
static __device__ __forceinline__ void async_ld_b128(unsigned lds,
                                                     unsigned long long src) {
  asm volatile("global_load_async_to_lds_b128 %0, %1, off"
               :: "v"(lds), "v"(src) : "memory");
}
static __device__ __forceinline__ void wait_async_0() {
  asm volatile("s_wait_asynccnt 0x0" ::: "memory");
}
static __device__ __forceinline__ void wait_async_3() {
  asm volatile("s_wait_asynccnt 0x3" ::: "memory");
}
static __device__ __forceinline__ void wait_async_5() {
  asm volatile("s_wait_asynccnt 0x5" ::: "memory");
}
// Generic (aperture) shared pointer -> wave-relative LDS byte offset.
static __device__ __forceinline__ unsigned lds_addr32(const void* p) {
  return (unsigned)(unsigned long long)p;
}

// ---------------------------- small utilities ------------------------------

static __device__ __forceinline__ unsigned int pk_bf16_f32(float a, float b) {
  __bf16 x = (__bf16)a, y = (__bf16)b;
  unsigned short ux, uy;
  __builtin_memcpy(&ux, &x, 2);
  __builtin_memcpy(&uy, &y, 2);
  return (unsigned int)ux | ((unsigned int)uy << 16);
}

__global__ void zero_f32(float* __restrict__ p, long n) {
  long i = (long)blockIdx.x * blockDim.x + threadIdx.x;
  long s = (long)gridDim.x * blockDim.x;
  for (; i < n; i += s) p[i] = 0.f;
}

// f32 -> bf16 (row-major kept), 4 elements/thread
__global__ void cvt_bf16(const float* __restrict__ in, __bf16* __restrict__ out,
                         long n4) {
  long i = (long)blockIdx.x * blockDim.x + threadIdx.x;
  long s = (long)gridDim.x * blockDim.x;
  for (; i < n4; i += s) {
    float4 v = ((const float4*)in)[i];
    uint2 o;
    o.x = pk_bf16_f32(v.x, v.y);
    o.y = pk_bf16_f32(v.z, v.w);
    ((uint2*)out)[i] = o;
  }
}

// f32 [z][R][C] -> bf16 [z][C][R]  (transpose so GEMM B-tiles are k-contiguous)
__global__ __launch_bounds__(256) void cvt_bf16_T(const float* __restrict__ in,
                                                  __bf16* __restrict__ out,
                                                  int R, int C) {
  __shared__ float tile[32][33];
  int z = blockIdx.z;
  long ibase = (long)z * R * C;
  long obase = (long)z * R * C;
  int c0 = blockIdx.x * 32, r0 = blockIdx.y * 32;
  int j = threadIdx.x & 31;
  int i = threadIdx.x >> 5;
#pragma unroll
  for (int s = 0; s < 32; s += 8)
    tile[i + s][j] = in[ibase + (long)(r0 + i + s) * C + (c0 + j)];
  __syncthreads();
#pragma unroll
  for (int s = 0; s < 32; s += 8)
    out[obase + (long)(c0 + i + s) * R + (r0 + j)] = (__bf16)tile[j][i + s];
}

// ------------------------------ gate / top-2 -------------------------------
__global__ __launch_bounds__(256) void gate_topk(
    const float* __restrict__ x, const float* __restrict__ gw,
    const float* __restrict__ gb, int* __restrict__ counts,
    int* __restrict__ tok_idx, float* __restrict__ tok_w) {
  int lane = threadIdx.x & 31;
  int wave = threadIdx.x >> 5;
  int tok  = blockIdx.x * 8 + wave;
  if (tok >= NTOK) return;

  float acc[NEXP];
#pragma unroll
  for (int e = 0; e < NEXP; ++e) acc[e] = 0.f;

  const float* xr = x + (long)tok * DDIM;
  for (int d = lane; d < DDIM; d += 32) {
    float xv = xr[d];
    const float4* g = (const float4*)(gw + (long)d * NEXP);
#pragma unroll
    for (int q = 0; q < 4; ++q) {
      float4 gv = g[q];
      acc[q * 4 + 0] += xv * gv.x;
      acc[q * 4 + 1] += xv * gv.y;
      acc[q * 4 + 2] += xv * gv.z;
      acc[q * 4 + 3] += xv * gv.w;
    }
  }
#pragma unroll
  for (int e = 0; e < NEXP; ++e) {
#pragma unroll
    for (int off = 16; off > 0; off >>= 1)
      acc[e] += __shfl_xor(acc[e], off, 32);
  }

  if (lane == 0) {
    float m = acc[0];
#pragma unroll
    for (int e = 1; e < NEXP; ++e) m = fmaxf(m, acc[e]);
    float p[NEXP], s = 0.f;
#pragma unroll
    for (int e = 0; e < NEXP; ++e) { p[e] = __expf(acc[e] - m); s += p[e]; }
    float inv = 1.f / s;
    float sc[NEXP];
#pragma unroll
    for (int e = 0; e < NEXP; ++e) sc[e] = p[e] * inv + gb[e];

    int i1 = 0;
#pragma unroll
    for (int e = 1; e < NEXP; ++e) if (sc[e] > sc[i1]) i1 = e;
    int i2 = (i1 == 0) ? 1 : 0;
#pragma unroll
    for (int e = 0; e < NEXP; ++e)
      if (e != i1 && sc[e] > sc[i2]) i2 = e;

    int ids[2] = {i1, i2};
#pragma unroll
    for (int k = 0; k < 2; ++k) {
      int e = ids[k];
      float wgt = acc[e] * RSCALE;           // reference uses raw logits
      int slot = atomicAdd(&counts[e], 1);
      if (slot < CAP) {
        tok_idx[e * CAP + slot] = tok;
        tok_w[e * CAP + slot]   = wgt;
      }
    }
  }
}

// -------------------------- WMMA fragment helpers --------------------------
static __device__ __forceinline__ v16bf frag_A(const __bf16* lds_row, int lane) {
  const unsigned int* p = (const unsigned int*)lds_row;
  union { v16bf v; unsigned int u[8]; } f;
  int hi = (lane >> 4) & 1;
#pragma unroll
  for (int j = 0; j < 8; ++j)
    f.u[j] = p[((j < 4) ? 0 : 8) + hi * 4 + (j & 3)];
  return f.v;
}
static __device__ __forceinline__ v16bf frag_B(const __bf16* lds_col, int lane) {
  const unsigned int* p = (const unsigned int*)lds_col;
  union { v16bf v; unsigned int u[8]; } f;
  int hi = (lane >> 4) & 1;
#pragma unroll
  for (int j = 0; j < 8; ++j) f.u[j] = p[((j < 4) ? 0 : 8) + hi * 4 + (j & 3)];
  return f.v;
}

// ------------------------- GEMM1 + SwiGLU (fused) --------------------------
// 64(M) x 128(N) tile; 8 waves 2x4; 2x2 frags/wave for both paths ->
// 8 WMMA per 32-k step per wave. Double-buffered LDS fed by async DMA.
__global__ __launch_bounds__(256) void gemm1_swiglu(
    const __bf16* __restrict__ X,      // [*, DDIM]
    const __bf16* __restrict__ W1t,    // [z][HDIM][DDIM]
    const __bf16* __restrict__ W3t,
    __bf16* __restrict__ Hout,         // [z][rows_cap][HDIM]
    const int* __restrict__ tok_idx,   // nullable
    const int* __restrict__ counts,    // nullable
    int rows_cap,
    const float* __restrict__ zpage) { // 256B of zeros (invalid-row source)
  __shared__ __align__(16) __bf16 lA[2][64 * 32];
  __shared__ __align__(16) __bf16 lB1[2][128 * 32];
  __shared__ __align__(16) __bf16 lB3[2][128 * 32];
  __shared__ int sTok[64];

  const int z = blockIdx.z;
  const int n0 = blockIdx.x * 128;
  const int m0 = blockIdx.y * 64;
  const int tid = threadIdx.x;
  const int lane = tid & 31, wave = tid >> 5;
  const int wm = wave & 1, wn = wave >> 1;
  const int nrows = counts ? min(counts[z], rows_cap) : rows_cap;

  if (tid < 64) {
    int r = m0 + tid;
    sTok[tid] = (r < nrows) ? (tok_idx ? tok_idx[z * CAP + r] : r) : -1;
  }
  __syncthreads();

  const int rA = tid >> 2, cA = (tid & 3) * 8;    // A: 64 x 32, 16B/thread
  const int nB = tid >> 1, cB = (tid & 1) * 16;   // B: 128 x 32, 2x16B/thread
  const int tkA = sTok[rA];
  const __bf16* xrow  = (tkA >= 0) ? X + (long)tkA * DDIM : nullptr;
  const __bf16* b1row = W1t + (long)z * HDIM * DDIM + (long)(n0 + nB) * DDIM;
  const __bf16* b3row = W3t + (long)z * HDIM * DDIM + (long)(n0 + nB) * DDIM;

  // 5 async-copy instructions per wave per slab
  auto issueSlab = [&](int k0, int b) {
    const __bf16* asrc = xrow ? (xrow + k0 + cA) : (const __bf16*)zpage;
    async_ld_b128(lds_addr32(&lA[b][rA * 32 + cA]), (unsigned long long)asrc);
    async_ld_b128(lds_addr32(&lB1[b][nB * 32 + cB]),
                  (unsigned long long)(b1row + k0 + cB));
    async_ld_b128(lds_addr32(&lB1[b][nB * 32 + cB + 8]),
                  (unsigned long long)(b1row + k0 + cB + 8));
    async_ld_b128(lds_addr32(&lB3[b][nB * 32 + cB]),
                  (unsigned long long)(b3row + k0 + cB));
    async_ld_b128(lds_addr32(&lB3[b][nB * 32 + cB + 8]),
                  (unsigned long long)(b3row + k0 + cB + 8));
  };

  v8f acc1[2][2] = {}; v8f acc3[2][2] = {};

  issueSlab(0, 0);
  int buf = 0;
  for (int k0 = 0; k0 < DDIM; k0 += 32) {
    if (k0 + 32 < DDIM) {
      issueSlab(k0 + 32, buf ^ 1);  // overlap next DMA with this wait
      wait_async_5();               // 5 in flight = next slab; current done
    } else {
      wait_async_0();
    }
    __syncthreads();

    v16bf af[2], bf1[2], bf3[2];
#pragma unroll
    for (int mi = 0; mi < 2; ++mi)
      af[mi] = frag_A(&lA[buf][(wm * 32 + mi * 16 + (lane & 15)) * 32], lane);
#pragma unroll
    for (int ni = 0; ni < 2; ++ni) {
      bf1[ni] = frag_B(&lB1[buf][(wn * 32 + ni * 16 + (lane & 15)) * 32], lane);
      bf3[ni] = frag_B(&lB3[buf][(wn * 32 + ni * 16 + (lane & 15)) * 32], lane);
    }
#pragma unroll
    for (int mi = 0; mi < 2; ++mi)
#pragma unroll
      for (int ni = 0; ni < 2; ++ni) {
        acc1[mi][ni] = __builtin_amdgcn_wmma_f32_16x16x32_bf16(
            false, af[mi], false, bf1[ni], (short)0, acc1[mi][ni], false, false);
        acc3[mi][ni] = __builtin_amdgcn_wmma_f32_16x16x32_bf16(
            false, af[mi], false, bf3[ni], (short)0, acc3[mi][ni], false, false);
      }
    __syncthreads();  // compute done before this buffer is re-filled
    buf ^= 1;
  }

  const int hi = (lane >> 4) & 1;
  __bf16* hb = Hout + (long)z * rows_cap * HDIM;
#pragma unroll
  for (int mi = 0; mi < 2; ++mi) {
#pragma unroll
    for (int ni = 0; ni < 2; ++ni) {
      union { v8f v; float f[8]; } u1, u3;
      u1.v = acc1[mi][ni]; u3.v = acc3[mi][ni];
      int col = n0 + wn * 32 + ni * 16 + (lane & 15);
#pragma unroll
      for (int r = 0; r < 8; ++r) {
        int row = m0 + wm * 32 + mi * 16 + r + hi * 8;
        if (row < nrows) {
          float a1 = u1.f[r];
          float hv = (a1 / (1.f + __expf(-a1))) * u3.f[r];
          hb[(long)row * HDIM + col] = (__bf16)hv;
        }
      }
    }
  }
}

// ------------------------ GEMM2 + scaled scatter ---------------------------
__global__ __launch_bounds__(256) void gemm2_out(
    const __bf16* __restrict__ Hin,    // [z][rows_cap][HDIM]
    const __bf16* __restrict__ W2t,    // [z][DDIM][HDIM]
    float* __restrict__ Out,           // [NTOK][DDIM]
    const int* __restrict__ tok_idx,   // nullable
    const float* __restrict__ tok_w,   // nullable -> fixed_scale
    const int* __restrict__ counts,    // nullable
    int rows_cap, float fixed_scale,
    const float* __restrict__ zpage) {
  __shared__ __align__(16) __bf16 lA[2][64 * 32];
  __shared__ __align__(16) __bf16 lB[2][128 * 32];
  __shared__ int   sTok[64];
  __shared__ float sWt[64];

  const int z = blockIdx.z;
  const int n0 = blockIdx.x * 128;
  const int m0 = blockIdx.y * 64;
  const int tid = threadIdx.x;
  const int lane = tid & 31, wave = tid >> 5;
  const int wm = wave & 1, wn = wave >> 1;
  const int nrows = counts ? min(counts[z], rows_cap) : rows_cap;

  if (tid < 64) {
    int r = m0 + tid;
    int tk = -1; float wt = fixed_scale;
    if (r < nrows) {
      tk = tok_idx ? tok_idx[z * CAP + r] : r;
      if (tok_w) wt = tok_w[z * CAP + r];
    }
    sTok[tid] = tk; sWt[tid] = wt;
  }
  __syncthreads();

  const int rA = tid >> 2, cA = (tid & 3) * 8;
  const int nB = tid >> 1, cB = (tid & 1) * 16;
  const __bf16* hrow = (m0 + rA < nrows)
                           ? Hin + ((long)z * rows_cap + m0 + rA) * HDIM
                           : nullptr;
  const __bf16* brow = W2t + (long)z * DDIM * HDIM + (long)(n0 + nB) * HDIM;

  // 3 async-copy instructions per wave per slab
  auto issueSlab = [&](int k0, int b) {
    const __bf16* asrc = hrow ? (hrow + k0 + cA) : (const __bf16*)zpage;
    async_ld_b128(lds_addr32(&lA[b][rA * 32 + cA]), (unsigned long long)asrc);
    async_ld_b128(lds_addr32(&lB[b][nB * 32 + cB]),
                  (unsigned long long)(brow + k0 + cB));
    async_ld_b128(lds_addr32(&lB[b][nB * 32 + cB + 8]),
                  (unsigned long long)(brow + k0 + cB + 8));
  };

  v8f acc[2][2] = {};

  issueSlab(0, 0);
  int buf = 0;
  for (int k0 = 0; k0 < HDIM; k0 += 32) {
    if (k0 + 32 < HDIM) {
      issueSlab(k0 + 32, buf ^ 1);
      wait_async_3();
    } else {
      wait_async_0();
    }
    __syncthreads();

    v16bf af[2], bf[2];
#pragma unroll
    for (int mi = 0; mi < 2; ++mi)
      af[mi] = frag_A(&lA[buf][(wm * 32 + mi * 16 + (lane & 15)) * 32], lane);
#pragma unroll
    for (int ni = 0; ni < 2; ++ni)
      bf[ni] = frag_B(&lB[buf][(wn * 32 + ni * 16 + (lane & 15)) * 32], lane);
#pragma unroll
    for (int mi = 0; mi < 2; ++mi)
#pragma unroll
      for (int ni = 0; ni < 2; ++ni)
        acc[mi][ni] = __builtin_amdgcn_wmma_f32_16x16x32_bf16(
            false, af[mi], false, bf[ni], (short)0, acc[mi][ni], false, false);
    __syncthreads();
    buf ^= 1;
  }

  const int hi = (lane >> 4) & 1;
#pragma unroll
  for (int mi = 0; mi < 2; ++mi) {
#pragma unroll
    for (int ni = 0; ni < 2; ++ni) {
      union { v8f v; float f[8]; } ua;
      ua.v = acc[mi][ni];
      int col = n0 + wn * 32 + ni * 16 + (lane & 15);
#pragma unroll
      for (int r = 0; r < 8; ++r) {
        int lrow = wm * 32 + mi * 16 + r + hi * 8;
        int row = m0 + lrow;
        if (row < nrows) {
          int tk = sTok[lrow];
          float wt = sWt[lrow];
          if (tk >= 0)
            atomicAdd(&Out[(long)tk * DDIM + col], ua.f[r] * wt);
        }
      }
    }
  }
}

// ------------------------------- launcher ----------------------------------
extern "C" void kernel_launch(void* const* d_in, const int* in_sizes, int n_in,
                              void* d_out, int out_size, void* d_ws,
                              size_t ws_size, hipStream_t stream) {
  const float* x      = (const float*)d_in[0];
  const float* gate_w = (const float*)d_in[1];
  const float* gate_b = (const float*)d_in[2];
  const float* ew1    = (const float*)d_in[3];
  const float* ew3    = (const float*)d_in[4];
  const float* ew2    = (const float*)d_in[5];
  const float* sw1    = (const float*)d_in[6];
  const float* sw3    = (const float*)d_in[7];
  const float* sw2    = (const float*)d_in[8];
  float* out = (float*)d_out;

  // -------- workspace layout (~429 MB) --------
  char* ws = (char*)d_ws;
  size_t off = 0;
  auto take = [&](size_t bytes) -> void* {
    void* p = ws + off;
    off = (off + bytes + 255) & ~(size_t)255;
    return p;
  };
  int*    counts  = (int*)take(NEXP * sizeof(int));
  float*  zpage   = (float*)take(256);
  int*    tok_idx = (int*)take((size_t)NEXP * CAP * sizeof(int));
  float*  tok_w   = (float*)take((size_t)NEXP * CAP * sizeof(float));
  __bf16* x_bf    = (__bf16*)take((size_t)NTOK * DDIM * 2);
  __bf16* w1_bf   = (__bf16*)take((size_t)NEXP * DDIM * HDIM * 2);  // [H][D]
  __bf16* w3_bf   = (__bf16*)take((size_t)NEXP * DDIM * HDIM * 2);  // [H][D]
  __bf16* w2_bf   = (__bf16*)take((size_t)NEXP * HDIM * DDIM * 2);  // [D][H]
  __bf16* s1_bf   = (__bf16*)take((size_t)NSH * DDIM * HDIM * 2);
  __bf16* s3_bf   = (__bf16*)take((size_t)NSH * DDIM * HDIM * 2);
  __bf16* s2_bf   = (__bf16*)take((size_t)NSH * HDIM * DDIM * 2);
  __bf16* h_buf   = (__bf16*)take((size_t)NEXP * CAP * HDIM * 2);   // reused

  // -------- phase 0: zero output, counters, zero-page --------
  zero_f32<<<2048, 256, 0, stream>>>(out, (long)NTOK * DDIM);
  zero_f32<<<1, 32, 0, stream>>>((float*)counts, NEXP);
  zero_f32<<<1, 64, 0, stream>>>(zpage, 64);

  // -------- phase 1: f32 -> bf16 (+ transpose weights to [n][k]) --------
  cvt_bf16<<<2048, 256, 0, stream>>>(x, x_bf, (long)NTOK * DDIM / 4);
  dim3 t13(HDIM / 32, DDIM / 32, NEXP);   // [D][H] -> [H][D]
  cvt_bf16_T<<<t13, 256, 0, stream>>>(ew1, w1_bf, DDIM, HDIM);
  cvt_bf16_T<<<t13, 256, 0, stream>>>(ew3, w3_bf, DDIM, HDIM);
  dim3 t2(DDIM / 32, HDIM / 32, NEXP);    // [H][D] -> [D][H]
  cvt_bf16_T<<<t2, 256, 0, stream>>>(ew2, w2_bf, HDIM, DDIM);
  dim3 ts13(HDIM / 32, DDIM / 32, NSH);
  cvt_bf16_T<<<ts13, 256, 0, stream>>>(sw1, s1_bf, DDIM, HDIM);
  cvt_bf16_T<<<ts13, 256, 0, stream>>>(sw3, s3_bf, DDIM, HDIM);
  dim3 ts2(DDIM / 32, HDIM / 32, NSH);
  cvt_bf16_T<<<ts2, 256, 0, stream>>>(sw2, s2_bf, HDIM, DDIM);

  // -------- phase 2: gate + top-2 routing --------
  gate_topk<<<NTOK / 8, 256, 0, stream>>>(x, gate_w, gate_b, counts, tok_idx,
                                          tok_w);

  // -------- phase 3: routed experts --------
  dim3 g1r(HDIM / 128, CAP / 64, NEXP);
  gemm1_swiglu<<<g1r, 256, 0, stream>>>(x_bf, w1_bf, w3_bf, h_buf, tok_idx,
                                        counts, CAP, zpage);
  dim3 g2r(DDIM / 128, CAP / 64, NEXP);
  gemm2_out<<<g2r, 256, 0, stream>>>(h_buf, w2_bf, out, tok_idx, tok_w, counts,
                                     CAP, 1.0f, zpage);

  // -------- phase 4: shared experts (reuse h_buf; stream-ordered) --------
  dim3 g1s(HDIM / 128, NTOK / 64, NSH);
  gemm1_swiglu<<<g1s, 256, 0, stream>>>(x_bf, s1_bf, s3_bf, h_buf,
                                        (const int*)nullptr,
                                        (const int*)nullptr, NTOK, zpage);
  dim3 g2s(DDIM / 128, NTOK / 64, NSH);
  gemm2_out<<<g2s, 256, 0, stream>>>(h_buf, s2_bf, out, (const int*)nullptr,
                                     (const float*)nullptr,
                                     (const int*)nullptr, NTOK, 1.0f / NSH,
                                     zpage);
}
